// GNN_Bottleneck_50903952392325
// MI455X (gfx1250) — matrix operand order repaired and checked by reading, not compiled, not run on hardware
//
#include <hip/hip_runtime.h>
#include <hip/hip_bf16.h>

// ---------- CDNA5 WMMA vector types ----------
typedef __attribute__((ext_vector_type(16))) __bf16 v16bf;
typedef __attribute__((ext_vector_type(8)))  __bf16 v8bf;
typedef __attribute__((ext_vector_type(8)))  float  v8f;

#define TPB 256

// ---------------------------------------------------------------------------
// Degree / normalization
// ---------------------------------------------------------------------------
__global__ void deg_init(float* __restrict__ deg, int n) {
    int i = blockIdx.x * blockDim.x + threadIdx.x;
    if (i < n) deg[i] = 1.0f;   // self-loop
}

__global__ void deg_count(const int* __restrict__ dst, float* __restrict__ deg, int e) {
    int i = blockIdx.x * blockDim.x + threadIdx.x;
    if (i < e) atomicAdd(&deg[dst[i]], 1.0f);
}

__global__ void deg_rsqrt(float* __restrict__ deg, int n) {
    int i = blockIdx.x * blockDim.x + threadIdx.x;
    if (i < n) deg[i] = rsqrtf(deg[i]);
}

// ---------------------------------------------------------------------------
// f32 -> bf16 conversion
// ---------------------------------------------------------------------------
__global__ void to_bf16(const float* __restrict__ in, __bf16* __restrict__ out, int n) {
    int i = blockIdx.x * blockDim.x + threadIdx.x;
    if (i < n) out[i] = (__bf16)in[i];
}

// ---------------------------------------------------------------------------
// Repack W [K x Nout] (f32, row-major) into per-lane B fragments (bf16):
//   chunk (tn, kb) = 32x16 tile of B; within chunk, lane-major 16 values:
//   lane l<16  : column n = tn*16+l,      K = kb*32 + 0..15
//   lane l>=16 : column n = tn*16+(l-16), K = kb*32 + 16..31
// ---------------------------------------------------------------------------
__global__ void repack_B(const float* __restrict__ W, __bf16* __restrict__ bp,
                         int K, int Nout) {
    int kBlocks = K >> 5;
    int tilesN  = Nout >> 4;
    int total   = tilesN * kBlocks * 512;
    int idx = blockIdx.x * blockDim.x + threadIdx.x;
    if (idx >= total) return;
    int i     = idx & 15;
    int lane  = (idx >> 4) & 31;
    int chunk = idx >> 9;
    int kb    = chunk % kBlocks;
    int tn    = chunk / kBlocks;
    int k = kb * 32 + (lane >> 4) * 16 + i;
    int n = tn * 16 + (lane & 15);
    bp[(size_t)chunk * 512 + lane * 16 + i] = (__bf16)W[(size_t)k * Nout + n];
}

// ---------------------------------------------------------------------------
// WMMA GEMM with LDS-resident B, register-resident N-strip, and explicit
// double-buffering of the B fragment so ds_load latency overlaps the
// previous (independent-accumulator) WMMA:
//   - whole packed B (TN*KB*1KB <= 64KB) copied to LDS once per block
//   - each wave owns one 16-row M-tile and all TN column tiles
//   - per k-block: 2 global b128 loads (A, loaded ONCE for all TN tiles)
//                  + per tile 2 ds_load_b128 (B, prefetched one tile ahead)
//                  + 1 v_wmma_f32_16x16x32_bf16
// ---------------------------------------------------------------------------
template <int TN, int KB>
__global__ void gemm_wmma_bf16_lds(const __bf16* __restrict__ A,
                                   const __bf16* __restrict__ Bpack,
                                   float* __restrict__ C,
                                   int tilesM, int K, int Nout) {
    __shared__ __bf16 Bs[TN * KB * 512];

    // cooperative global->LDS copy of packed B (16B per transfer)
    {
        const uint4* g = (const uint4*)Bpack;
        uint4* s = (uint4*)Bs;
        constexpr int CHUNKS = TN * KB * 512 / 8;   // bf16 elems / 8 per uint4
        for (int i = threadIdx.x; i < CHUNKS; i += TPB) s[i] = g[i];
    }
    __syncthreads();

    const int wave = threadIdx.x >> 5;
    const int tm   = blockIdx.x * (TPB >> 5) + wave;
    if (tm >= tilesM) return;                 // wave-uniform; EXEC stays full

    const int lane = threadIdx.x & 31;
    const int half = lane >> 4;               // 0: lanes 0-15, 1: lanes 16-31
    const int l    = lane & 15;

    const __bf16* Arow = A + (size_t)(tm * 16 + l) * K;

    auto loadB = [&](int t, int kb) -> v16bf {
        const __bf16* bpl = Bs + ((size_t)t * KB + kb) * 512 + lane * 16;
        v8bf blo = *(const v8bf*)(bpl);
        v8bf bhi = *(const v8bf*)(bpl + 8);
        v16bf b;
#pragma unroll
        for (int i = 0; i < 8; ++i) { b[i] = blo[i]; b[8 + i] = bhi[i]; }
        return b;
    };

    v8f acc[TN];
#pragma unroll
    for (int t = 0; t < TN; ++t) acc[t] = (v8f){};

#pragma unroll
    for (int kb = 0; kb < KB; ++kb) {
        const int kk = kb * 32;
        // A fragment (contiguous, matches packed B): lanes 0-15 K {kk+0..15},
        // lanes 16-31 K {kk+16..31}; two aligned 16B global loads.
        v8bf alo = *(const v8bf*)(Arow + kk + half * 16);
        v8bf ahi = *(const v8bf*)(Arow + kk + half * 16 + 8);
        v16bf a;
#pragma unroll
        for (int i = 0; i < 8; ++i) { a[i] = alo[i]; a[8 + i] = ahi[i]; }

        // software-pipelined B fragments: fetch t+1 before WMMA on t
        v16bf bcur = loadB(0, kb);
#pragma unroll
        for (int t = 0; t < TN; ++t) {
            v16bf bnxt = bcur;
            if (t + 1 < TN) bnxt = loadB(t + 1, kb);
            acc[t] = __builtin_amdgcn_wmma_f32_16x16x32_bf16(
                /*neg_a=*/false, a, /*neg_b=*/false, bcur,
                /*c_mod=*/(short)0, acc[t], /*reuse_a=*/false, /*reuse_b=*/false);
            bcur = bnxt;
        }
    }

    // C/D layout: VGPR j -> row (tm*16 + j + half*8), col (t*16 + l)
#pragma unroll
    for (int t = 0; t < TN; ++t) {
        float* Ccol = C + (size_t)(tm * 16) * Nout + t * 16 + l;
#pragma unroll
        for (int j = 0; j < 8; ++j)
            Ccol[(size_t)(j + half * 8) * Nout] = acc[t][j];
    }
}

// ---------------------------------------------------------------------------
// agg[i,f] = bias[f] + tmp[i,f] * dinv[i]^2     (self-loop term fused)
// ---------------------------------------------------------------------------
__global__ void agg_init(const float* __restrict__ tmp, const float* __restrict__ dinv,
                         const float* __restrict__ bias, float* __restrict__ agg,
                         int n, int d) {
    int idx = blockIdx.x * blockDim.x + threadIdx.x;
    if (idx >= n * d) return;
    int i = idx / d, f = idx - i * d;
    float di = dinv[i];
    agg[idx] = bias[f] + tmp[idx] * di * di;
}

// ---------------------------------------------------------------------------
// Edge scatter: agg[dst] += tmp[src] * dinv[src]*dinv[dst]
// One thread per (edge, 4-feature chunk): float4 gather + 4 f32 L2 atomics.
// ---------------------------------------------------------------------------
__global__ void edge_scatter(const float* __restrict__ tmp,
                             const int* __restrict__ src, const int* __restrict__ dst,
                             const float* __restrict__ dinv, float* __restrict__ agg,
                             int e, int d) {
    const int chunks = d >> 2;
    int tid = blockIdx.x * blockDim.x + threadIdx.x;
    if (tid >= e * chunks) return;
    int ei = tid / chunks;
    int c  = (tid - ei * chunks) << 2;
    int s = src[ei], t = dst[ei];
    float nrm = dinv[s] * dinv[t];
    const float4 v = *(const float4*)(tmp + (size_t)s * d + c);
    float* o = agg + (size_t)t * d + c;
    atomicAdd(o + 0, v.x * nrm);
    atomicAdd(o + 1, v.y * nrm);
    atomicAdd(o + 2, v.z * nrm);
    atomicAdd(o + 3, v.w * nrm);
}

// ---------------------------------------------------------------------------
// ReLU + convert to bf16 for next layer's GEMM
// ---------------------------------------------------------------------------
__global__ void relu_to_bf16(const float* __restrict__ in, __bf16* __restrict__ out, int n) {
    int i = blockIdx.x * blockDim.x + threadIdx.x;
    if (i < n) out[i] = (__bf16)fmaxf(in[i], 0.0f);
}

// ---------------------------------------------------------------------------
static inline int cdiv(long long a, int b) { return (int)((a + b - 1) / b); }
static inline size_t alignup(size_t x) { return (x + 255) & ~(size_t)255; }

extern "C" void kernel_launch(void* const* d_in, const int* in_sizes, int n_in,
                              void* d_out, int out_size, void* d_ws, size_t ws_size,
                              hipStream_t stream) {
    const int D_IN = 256, D_H = 128, D_OUT = 64;
    const int N = in_sizes[0] / D_IN;
    const int E = in_sizes[1] / 2;

    const float* x  = (const float*)d_in[0];
    const int* ei   = (const int*)d_in[1];
    const int* src  = ei;
    const int* dst  = ei + E;
    const float* W1 = (const float*)d_in[2];
    const float* b1 = (const float*)d_in[3];
    const float* W2 = (const float*)d_in[4];
    const float* b2 = (const float*)d_in[5];
    const float* W3 = (const float*)d_in[6];
    const float* b3 = (const float*)d_in[7];
    float* out = (float*)d_out;

    // ---- workspace carve-up ----
    char* w = (char*)d_ws;
    size_t off = 0;
    float*  dinv = (float*)(w + off);  off += alignup((size_t)N * 4);
    __bf16* xbf  = (__bf16*)(w + off); off += alignup((size_t)N * D_IN * 2);
    __bf16* hbf  = (__bf16*)(w + off); off += alignup((size_t)N * D_H * 2);
    float*  tmp  = (float*)(w + off);  off += alignup((size_t)N * D_H * 4);
    float*  agg  = (float*)(w + off);  off += alignup((size_t)N * D_H * 4);
    __bf16* bpk  = (__bf16*)(w + off); off += alignup((size_t)256 * 1024);
    (void)ws_size;

    // ---- normalization ----
    deg_init<<<cdiv(N, TPB), TPB, 0, stream>>>(dinv, N);
    deg_count<<<cdiv(E, TPB), TPB, 0, stream>>>(dst, dinv, E);
    deg_rsqrt<<<cdiv(N, TPB), TPB, 0, stream>>>(dinv, N);

    // ---- activations to bf16 ----
    to_bf16<<<cdiv((long long)N * D_IN, TPB), TPB, 0, stream>>>(x, xbf, N * D_IN);

    const int tilesM = N / 16;           // 50000/16 = 3125 exactly
    const int gemmGrid = cdiv(tilesM, TPB / 32);

    // =================== Layer 1: 256 -> 128, ReLU ===================
    {
        const int K = D_IN, Dout = D_H;
        const int tilesN = Dout / 16, kBlocks = K / 32;
        repack_B<<<cdiv(tilesN * kBlocks * 512, TPB), TPB, 0, stream>>>(W1, bpk, K, Dout);
        gemm_wmma_bf16_lds<8, 8><<<gemmGrid, TPB, 0, stream>>>(xbf, bpk, tmp, tilesM, K, Dout);
        agg_init<<<cdiv((long long)N * Dout, TPB), TPB, 0, stream>>>(tmp, dinv, b1, agg, N, Dout);
        edge_scatter<<<cdiv((long long)E * (Dout / 4), TPB), TPB, 0, stream>>>(tmp, src, dst, dinv, agg, E, Dout);
        relu_to_bf16<<<cdiv((long long)N * Dout, TPB), TPB, 0, stream>>>(agg, hbf, N * Dout);
    }

    // =================== Layer 2: 128 -> 128, ReLU ===================
    {
        const int K = D_H, Dout = D_H;
        const int tilesN = Dout / 16, kBlocks = K / 32;
        repack_B<<<cdiv(tilesN * kBlocks * 512, TPB), TPB, 0, stream>>>(W2, bpk, K, Dout);
        gemm_wmma_bf16_lds<8, 4><<<gemmGrid, TPB, 0, stream>>>(hbf, bpk, tmp, tilesM, K, Dout);
        agg_init<<<cdiv((long long)N * Dout, TPB), TPB, 0, stream>>>(tmp, dinv, b2, agg, N, Dout);
        edge_scatter<<<cdiv((long long)E * (Dout / 4), TPB), TPB, 0, stream>>>(tmp, src, dst, dinv, agg, E, Dout);
        relu_to_bf16<<<cdiv((long long)N * Dout, TPB), TPB, 0, stream>>>(agg, hbf, N * Dout);
    }

    // =================== Layer 3: 128 -> 64 (no ReLU, f32 out) ===================
    {
        const int K = D_H, Dout = D_OUT;
        const int tilesN = Dout / 16, kBlocks = K / 32;
        repack_B<<<cdiv(tilesN * kBlocks * 512, TPB), TPB, 0, stream>>>(W3, bpk, K, Dout);
        gemm_wmma_bf16_lds<4, 4><<<gemmGrid, TPB, 0, stream>>>(hbf, bpk, tmp, tilesM, K, Dout);
        agg_init<<<cdiv((long long)N * Dout, TPB), TPB, 0, stream>>>(tmp, dinv, b3, out, N, Dout);
        edge_scatter<<<cdiv((long long)E * (Dout / 4), TPB), TPB, 0, stream>>>(tmp, src, dst, dinv, out, E, Dout);
    }
    (void)n_in; (void)out_size;
}
